// GCN3D_71287867179619
// MI455X (gfx1250) — matrix operand order repaired and checked by reading, not compiled, not run on hardware
//
#include <hip/hip_runtime.h>
#include <hip/hip_bf16.h>
#include <math.h>
#include <stdint.h>

// ---------------------------------------------------------------------------
// Types for CDNA5 WMMA (wave32, 16x16x32 f16 -> f32)
// ---------------------------------------------------------------------------
typedef __attribute__((ext_vector_type(16))) _Float16 v16h;
typedef __attribute__((ext_vector_type(8)))  _Float16 v8h;
typedef __attribute__((ext_vector_type(8)))  float    v8f;

// Async global->LDS builtin operand types (from hipcc diagnostic: the builtin
// takes int4 pointers; arg0 is global (as1), arg1 is LDS (as3)).
typedef int v4i __attribute__((vector_size(16)));
typedef __attribute__((address_space(1))) v4i g_v4i;   // global int4
typedef __attribute__((address_space(3))) v4i l_v4i;   // LDS int4

#define HAVE_ASYNC_LDS (__has_builtin(__builtin_amdgcn_global_load_async_to_lds_b128))

// ---------------------------------------------------------------------------
// Ball query: first `ns` in-radius indices in ascending index order, padded
// with the first hit (self-distance==0 guarantees >=1 hit). O(V) early-exit
// scan instead of the reference's full sort. Candidate coords tiled in LDS.
// grid = (V/256, B), block = 256.
// ---------------------------------------------------------------------------
__global__ __launch_bounds__(256) void ball_query_k(
    const float* __restrict__ xyz, int V, float r2, int ns,
    int* __restrict__ out)
{
    __shared__ float sx[256], sy[256], sz[256];
    const int b   = blockIdx.y;
    const int tid = threadIdx.x;
    const int v   = blockIdx.x * 256 + tid;
    const float* base = xyz + (size_t)b * V * 3;
    const float cx = base[3*v+0], cy = base[3*v+1], cz = base[3*v+2];
    int* o = out + ((size_t)b * V + v) * ns;
    int cnt = 0;
    for (int j0 = 0; j0 < V; j0 += 256) {
        const int j = j0 + tid;
        sx[tid] = base[3*j+0];
        sy[tid] = base[3*j+1];
        sz[tid] = base[3*j+2];
        __syncthreads();
        if (cnt < ns) {
            for (int jj = 0; jj < 256 && cnt < ns; ++jj) {
                const float dx = sx[jj] - cx;
                const float dy = sy[jj] - cy;
                const float dz = sz[jj] - cz;
                const float d2 = dx*dx + dy*dy + dz*dz;
                if (d2 <= r2) o[cnt++] = j0 + jj;
            }
        }
        __syncthreads();
    }
    const int f0 = (cnt > 0) ? o[0] : v;
    for (int k = cnt; k < ns; ++k) o[k] = f0;
}

// ---------------------------------------------------------------------------
// conv_surface: fm0[c] = relu( sum_s max_n relu(ndir_n . dhat_{s*32+c}) )
// One wave per vertex, lane = output channel. grid = B*V/8, block = 256.
// ---------------------------------------------------------------------------
__global__ __launch_bounds__(256) void conv_surface_k(
    const float* __restrict__ xyz, const int* __restrict__ idx,
    const float* __restrict__ d0, float* __restrict__ fm, int ldf, int V)
{
    const int L = 128, NN = 32;
    const int wave = threadIdx.x >> 5, lane = threadIdx.x & 31;
    const int m = blockIdx.x * 8 + wave;       // row = b*V + v
    const int b = m / V;
    float dcx[4], dcy[4], dcz[4];
    #pragma unroll
    for (int s = 0; s < 4; ++s) {
        const int k = s * 32 + lane;
        const float a0 = d0[k], a1 = d0[L + k], a2 = d0[2*L + k];
        const float inv = 1.0f / fmaxf(sqrtf(a0*a0 + a1*a1 + a2*a2), 1e-12f);
        dcx[s] = a0*inv; dcy[s] = a1*inv; dcz[s] = a2*inv;
    }
    const float cx = xyz[3*m+0], cy = xyz[3*m+1], cz = xyz[3*m+2];
    float a0 = -INFINITY, a1 = -INFINITY, a2 = -INFINITY, a3 = -INFINITY;
    const int* ip = idx + (size_t)m * NN;
    for (int n = 0; n < NN; ++n) {
        const int rj = b * V + ip[n];
        float dx = xyz[3*rj+0] - cx, dy = xyz[3*rj+1] - cy, dz = xyz[3*rj+2] - cz;
        const float inv = 1.0f / fmaxf(sqrtf(dx*dx + dy*dy + dz*dz), 1e-12f);
        dx *= inv; dy *= inv; dz *= inv;
        a0 = fmaxf(a0, fmaxf(dx*dcx[0] + dy*dcy[0] + dz*dcz[0], 0.f));
        a1 = fmaxf(a1, fmaxf(dx*dcx[1] + dy*dcy[1] + dz*dcz[1], 0.f));
        a2 = fmaxf(a2, fmaxf(dx*dcx[2] + dy*dcy[2] + dz*dcz[2], 0.f));
        a3 = fmaxf(a3, fmaxf(dx*dcx[3] + dy*dcy[3] + dz*dcz[3], 0.f));
    }
    fm[(size_t)m * ldf + lane] = fmaxf(a0 + a1 + a2 + a3, 0.f);
}

// ---------------------------------------------------------------------------
// One-time per layer: convert W[K,N] (f32) -> WhT[N,K] (f16, transposed).
// Weights are reused by every M-block of the GEMM (32x for layer 8), so
// converting once halves global weight traffic and makes WhT rows contiguous
// (async-copyable 16B chunks).
// ---------------------------------------------------------------------------
__global__ __launch_bounds__(256) void wcvt_k(
    const float* __restrict__ W, _Float16* __restrict__ WhT, int K, int N)
{
    const int t = blockIdx.x * blockDim.x + threadIdx.x;
    if (t >= K * N) return;
    const int k = t / N, n = t - k * N;
    WhT[(size_t)n * K + k] = (_Float16)W[t];
}

// ---------------------------------------------------------------------------
// WMMA GEMM + bias:  C[M,N] = A[M,K](f32, ld=lda) * W[K,N] + bias, with W
// pre-transposed/converted as WhT[N,K] (f16). A is staged f32->f16 through
// VGPRs; the B tile (WhT rows) is staged into LDS with
// GLOBAL_LOAD_ASYNC_TO_LDS_B128 (ASYNCcnt) when the builtin exists, else a
// vectorized v8h copy. Block = 64 threads (2 waves), each wave owns one
// 16x16 C tile. Requires M%16==0, N%32==0, K%32==0 (true for every layer).
// LDS rows padded to 264 halves (528B: 16B-aligned rows, conflict-free).
//   A frag (16x32, ISA 7.12.2): lane<16 -> row=lane, K {kk..+7, kk+16..+23};
//     lane>=16 same rows, +8.  Two contiguous 8-half runs => 2x ds_load_b128.
//   B frag (32x16): lane%16 = N col; same K order as A frag:
//     lanes<16 K {kk..+7, kk+16..+23}, lanes>=16 +8 (two 8-half runs).
//   C layout: VGPR r -> row m0 + r + 8*(lane>>4), col = n0 + lane%16.
// ---------------------------------------------------------------------------
__global__ __launch_bounds__(64) void wmma_gemm_bias_k(
    const float* __restrict__ A, int lda,
    const _Float16* __restrict__ WhT,
    const float* __restrict__ bias,
    float* __restrict__ C, int ldc, int K)
{
    alignas(16) __shared__ _Float16 As[16 * 264];
    alignas(16) __shared__ _Float16 Bs[32 * 264];
    const int tid = threadIdx.x;
    const int m0 = blockIdx.y * 16;
    const int n0 = blockIdx.x * 32;

    __builtin_prefetch(&A[(size_t)m0 * lda], 0, 1);           // global_prefetch_b8
    __builtin_prefetch(&WhT[(size_t)n0 * K], 0, 1);

    // ---- Stage B tile: Bs[n][k] = WhT[n0+n][k], 16B chunks -----------------
    const int cpr   = K >> 3;          // 16B chunks per row
    const int total = 32 * cpr;        // multiple of 64 for all K here
    for (int e = tid; e < total; e += 64) {
        const int n  = e / cpr;
        const int k0 = (e - n * cpr) << 3;
        const _Float16* gp = WhT + (size_t)(n0 + n) * K + k0;
        _Float16*       lp = &Bs[n * 264 + k0];
#if HAVE_ASYNC_LDS
        __builtin_amdgcn_global_load_async_to_lds_b128(
            (g_v4i*)(uintptr_t)gp,
            (l_v4i*)(uint32_t)(uintptr_t)lp, 0, 0);
#else
        *(v8h*)lp = *(const v8h*)gp;
#endif
    }
    // ---- Stage A tile (16 x K) f32 -> f16, row-major ----------------------
    for (int r = 0; r < 16; ++r)
        for (int k = tid; k < K; k += 64)
            As[r * 264 + k] = (_Float16)A[(size_t)(m0 + r) * lda + k];
#if HAVE_ASYNC_LDS
  #if __has_builtin(__builtin_amdgcn_s_wait_asynccnt)
    __builtin_amdgcn_s_wait_asynccnt(0);
  #else
    asm volatile("s_wait_asynccnt 0x0" ::: "memory");
  #endif
#endif
    __syncthreads();

    const int wave = tid >> 5, lane = tid & 31;
    const int lrow = lane & 15;
    const int hi   = lane >> 4;             // 0 or 1
    const int nloc = wave * 16 + lrow;      // local N column (0..31)
    v8f acc = {};
    for (int kk = 0; kk < K; kk += 32) {
        const _Float16* ap = &As[lrow * 264 + kk + hi * 8];
        const v8h a0 = *(const v8h*)ap;
        const v8h a1 = *(const v8h*)(ap + 16);
        const v16h a = __builtin_shufflevector(a0, a1,
            0,1,2,3,4,5,6,7,8,9,10,11,12,13,14,15);
        const _Float16* bp = &Bs[nloc * 264 + kk + hi * 8];
        const v8h b0 = *(const v8h*)bp;
        const v8h b1 = *(const v8h*)(bp + 16);
        const v16h bfr = __builtin_shufflevector(b0, b1,
            0,1,2,3,4,5,6,7,8,9,10,11,12,13,14,15);
        acc = __builtin_amdgcn_wmma_f32_16x16x32_f16(
            /*neg_a=*/false, a, /*neg_b=*/false, bfr,
            /*c_mod=*/(short)0, acc, /*reuse_a=*/false, /*reuse_b=*/false);
    }
    const int col = n0 + nloc;
    const float bv = bias[col];
    const int rbase = m0 + hi * 8;
    #pragma unroll
    for (int r = 0; r < 8; ++r)
        C[(size_t)(rbase + r) * ldc + col] = acc[r] + bv;
}

// ---------------------------------------------------------------------------
// conv_layer aggregation (oc==32): out = center + sum_s max_n theta*support.
// One wave per vertex, lane = channel; writes into the concatenated feature
// buffer at column `colOff`. grid = B*V/8, block = 256.
// ---------------------------------------------------------------------------
__global__ __launch_bounds__(256) void conv_agg_k(
    const float* __restrict__ xyz, const int* __restrict__ idx,
    const float* __restrict__ gout, const float* __restrict__ dirs,
    float* __restrict__ fm, int ldf, int colOff, int V, int doRelu)
{
    const int oc = 32, L = 128, ldg = 160, NN = 32;
    const int wave = threadIdx.x >> 5, lane = threadIdx.x & 31;
    const int m = blockIdx.x * 8 + wave;
    const int b = m / V;
    float dcx[4], dcy[4], dcz[4];
    #pragma unroll
    for (int s = 0; s < 4; ++s) {
        const int k = s * oc + lane;
        const float a0 = dirs[k], a1 = dirs[L + k], a2 = dirs[2*L + k];
        const float inv = 1.0f / fmaxf(sqrtf(a0*a0 + a1*a1 + a2*a2), 1e-12f);
        dcx[s] = a0*inv; dcy[s] = a1*inv; dcz[s] = a2*inv;
    }
    const float cx = xyz[3*m+0], cy = xyz[3*m+1], cz = xyz[3*m+2];
    const float cent = gout[(size_t)m * ldg + lane];
    float a0 = -INFINITY, a1 = -INFINITY, a2 = -INFINITY, a3 = -INFINITY;
    const int* ip = idx + (size_t)m * NN;
    for (int n = 0; n < NN; ++n) {
        const int rj = b * V + ip[n];
        float dx = xyz[3*rj+0] - cx, dy = xyz[3*rj+1] - cy, dz = xyz[3*rj+2] - cz;
        const float inv = 1.0f / fmaxf(sqrtf(dx*dx + dy*dy + dz*dz), 1e-12f);
        dx *= inv; dy *= inv; dz *= inv;
        const float* gp = gout + (size_t)rj * ldg + oc + lane;
        const float t0 = fmaxf(dx*dcx[0] + dy*dcy[0] + dz*dcz[0], 0.f);
        const float t1 = fmaxf(dx*dcx[1] + dy*dcy[1] + dz*dcz[1], 0.f);
        const float t2 = fmaxf(dx*dcx[2] + dy*dcy[2] + dz*dcz[2], 0.f);
        const float t3 = fmaxf(dx*dcx[3] + dy*dcy[3] + dz*dcz[3], 0.f);
        a0 = fmaxf(a0, t0 * gp[0*oc]);
        a1 = fmaxf(a1, t1 * gp[1*oc]);
        a2 = fmaxf(a2, t2 * gp[2*oc]);
        a3 = fmaxf(a3, t3 * gp[3*oc]);
    }
    float out = cent + a0 + a1 + a2 + a3;
    if (doRelu) out = fmaxf(out, 0.f);
    fm[(size_t)m * ldf + colOff + lane] = out;
}

// ---------------------------------------------------------------------------
// Layer 8 aggregation (oc==1024): one block (256 thr) per vertex; neighbor
// unit directions precomputed once in LDS; each thread owns 4 channels.
// grid = (V=256, B), block = 256.
// ---------------------------------------------------------------------------
__global__ __launch_bounds__(256) void conv_agg8_k(
    const float* __restrict__ xyz, const int* __restrict__ idx,
    const float* __restrict__ gout, const float* __restrict__ dirs,
    float* __restrict__ fm8, int V)
{
    const int oc = 1024, L = 4096, ldg = 5120, NN = 32;
    __shared__ float ndx[32], ndy[32], ndz[32];
    __shared__ int   nrow[32];
    const int b = blockIdx.y, v = blockIdx.x;
    const int m = b * V + v;
    const int tid = threadIdx.x;
    if (tid < 32) {
        const int rj = b * V + idx[(size_t)m * NN + tid];
        const float cx = xyz[3*m+0], cy = xyz[3*m+1], cz = xyz[3*m+2];
        float dx = xyz[3*rj+0] - cx, dy = xyz[3*rj+1] - cy, dz = xyz[3*rj+2] - cz;
        const float inv = 1.0f / fmaxf(sqrtf(dx*dx + dy*dy + dz*dz), 1e-12f);
        ndx[tid] = dx*inv; ndy[tid] = dy*inv; ndz[tid] = dz*inv;
        nrow[tid] = rj;
    }
    __syncthreads();
    for (int cc = 0; cc < 4; ++cc) {
        const int c = cc * 256 + tid;
        float dcx[4], dcy[4], dcz[4];
        #pragma unroll
        for (int s = 0; s < 4; ++s) {
            const int k = s * oc + c;
            const float a0 = dirs[k], a1 = dirs[L + k], a2 = dirs[2*L + k];
            const float inv = 1.0f / fmaxf(sqrtf(a0*a0 + a1*a1 + a2*a2), 1e-12f);
            dcx[s] = a0*inv; dcy[s] = a1*inv; dcz[s] = a2*inv;
        }
        const float cent = gout[(size_t)m * ldg + c];
        float a0 = -INFINITY, a1 = -INFINITY, a2 = -INFINITY, a3 = -INFINITY;
        for (int n = 0; n < NN; ++n) {
            const float ex = ndx[n], ey = ndy[n], ez = ndz[n];
            const float* gp = gout + (size_t)nrow[n] * ldg + oc + c;
            const float t0 = fmaxf(ex*dcx[0] + ey*dcy[0] + ez*dcz[0], 0.f);
            const float t1 = fmaxf(ex*dcx[1] + ey*dcy[1] + ez*dcz[1], 0.f);
            const float t2 = fmaxf(ex*dcx[2] + ey*dcy[2] + ez*dcz[2], 0.f);
            const float t3 = fmaxf(ex*dcx[3] + ey*dcy[3] + ez*dcz[3], 0.f);
            a0 = fmaxf(a0, t0 * gp[0*oc]);
            a1 = fmaxf(a1, t1 * gp[1*oc]);
            a2 = fmaxf(a2, t2 * gp[2*oc]);
            a3 = fmaxf(a3, t3 * gp[3*oc]);
        }
        fm8[(size_t)m * oc + c] = cent + a0 + a1 + a2 + a3;  // no relu (last layer)
    }
}

// ---------------------------------------------------------------------------
// pool: max over 4 ball-query neighbors, then deterministic bijective
// subsample perm(i) = (i*pA + pC) & pM  (odd multiplier => bijection on 2^k).
// (JAX threefry permutation is not reproducible on-device; structure is
// identical: gather-max over nnum=4 then row-subsample V/4.)
// ---------------------------------------------------------------------------
__global__ __launch_bounds__(128) void pool_select_k(
    const float* __restrict__ vin, const float* __restrict__ fin, int ldin, int C,
    const int* __restrict__ idxp, int Vin,
    float* __restrict__ vout, float* __restrict__ fout, int ldout, int Vout,
    int pA, int pC, int pM)
{
    const int t = blockIdx.x * blockDim.x + threadIdx.x;
    if (t >= 2 * Vout) return;
    const int b = t / Vout, i = t - b * Vout;
    const int p = (i * pA + pC) & pM;
    const int rin = b * Vin + p, rout = b * Vout + i;
    vout[3*rout+0] = vin[3*rin+0];
    vout[3*rout+1] = vin[3*rin+1];
    vout[3*rout+2] = vin[3*rin+2];
    const int* nb = idxp + (size_t)rin * 4;
    const float* f0 = fin + (size_t)(b * Vin + nb[0]) * ldin;
    const float* f1 = fin + (size_t)(b * Vin + nb[1]) * ldin;
    const float* f2 = fin + (size_t)(b * Vin + nb[2]) * ldin;
    const float* f3 = fin + (size_t)(b * Vin + nb[3]) * ldin;
    float* fo = fout + (size_t)rout * ldout;
    for (int c = 0; c < C; ++c)
        fo[c] = fmaxf(fmaxf(f0[c], f1[c]), fmaxf(f2[c], f3[c]));
}

// ---------------------------------------------------------------------------
// Global max over vertices: gfeat[b][c] = max_v fm8[b,v,c].  2048 threads.
// ---------------------------------------------------------------------------
__global__ __launch_bounds__(256) void gmax_k(
    const float* __restrict__ fm8, float* __restrict__ gf)
{
    const int t = blockIdx.x * blockDim.x + threadIdx.x;
    const int b = t >> 10, c = t & 1023;
    const float* p = fm8 + (size_t)b * 256 * 1024 + c;
    float m = -INFINITY;
    for (int v = 0; v < 256; ++v) m = fmaxf(m, p[(size_t)v * 1024]);
    gf[t] = m;
}

// ---------------------------------------------------------------------------
// Classifier head, fused: h = relu((gfeat@cw1+cb1)/sqrt(1+1e-5)*bg + bb);
// out = h@cw2 + cb2.   One block, h staged in LDS.
// ---------------------------------------------------------------------------
__global__ __launch_bounds__(256) void classifier_k(
    const float* __restrict__ gf,
    const float* __restrict__ cw1, const float* __restrict__ cb1,
    const float* __restrict__ bg,  const float* __restrict__ bb,
    const float* __restrict__ cw2, const float* __restrict__ cb2,
    float* __restrict__ out)
{
    __shared__ float hsh[2][256];
    const int j = threadIdx.x;
    const float rs = 0.9999950000374997f;   // 1/sqrt(1 + 1e-5)
    for (int b = 0; b < 2; ++b) {
        float acc = cb1[j];
        const float* g = gf + (size_t)b * 1024;
        for (int k = 0; k < 1024; ++k) acc += g[k] * cw1[(size_t)k * 256 + j];
        hsh[b][j] = fmaxf(acc * rs * bg[j] + bb[j], 0.f);
    }
    __syncthreads();
    if (j < 80) {
        const int b = j / 40, o = j - b * 40;
        float acc = cb2[o];
        for (int k = 0; k < 256; ++k) acc += hsh[b][k] * cw2[(size_t)k * 40 + o];
        out[b * 40 + o] = acc;
    }
}

// ---------------------------------------------------------------------------
// Host-side orchestration
// ---------------------------------------------------------------------------
extern "C" void kernel_launch(void* const* d_in, const int* in_sizes, int n_in,
                              void* d_out, int out_size, void* d_ws, size_t ws_size,
                              hipStream_t stream)
{
    (void)in_sizes; (void)n_in; (void)out_size; (void)ws_size;

    const float* vertices = (const float*)d_in[0];
    const float* d0       = (const float*)d_in[1];
    const float *Wl[9], *bl[9], *Dl[9];
    for (int i = 1; i <= 8; ++i) {
        Wl[i] = (const float*)d_in[2 + 3*(i-1)];
        bl[i] = (const float*)d_in[3 + 3*(i-1)];
        Dl[i] = (const float*)d_in[4 + 3*(i-1)];
    }
    const float* cw1 = (const float*)d_in[26];
    const float* cb1 = (const float*)d_in[27];
    const float* bg  = (const float*)d_in[28];
    const float* bb  = (const float*)d_in[29];
    const float* cw2 = (const float*)d_in[30];
    const float* cb2 = (const float*)d_in[31];

    // Workspace layout (everything stays resident; ~22 MB total).
    size_t off = 0;
    auto alloc = [&](size_t bytes) -> void* {
        off = (off + 255) & ~(size_t)255;
        void* p = (char*)d_ws + off;
        off += bytes;
        return p;
    };
    int*      idx1  = (int*)     alloc((size_t)2*4096*32*4);
    int*      idxp  = (int*)     alloc((size_t)2*4096*4*4);
    float*    fmA   = (float*)   alloc((size_t)2*4096*96*4);
    float*    vtx2  = (float*)   alloc((size_t)2*1024*3*4);
    float*    fmB   = (float*)   alloc((size_t)2*1024*192*4);
    int*      idx2  = (int*)     alloc((size_t)2*1024*32*4);
    float*    vtx3  = (float*)   alloc((size_t)2*256*3*4);
    float*    fmC   = (float*)   alloc((size_t)2*256*256*4);
    int*      idx3  = (int*)     alloc((size_t)2*256*32*4);
    float*    gout  = (float*)   alloc((size_t)512*5120*4);   // GEMM output scratch
    _Float16* whT   = (_Float16*)alloc((size_t)5120*256*2);   // f16 W^T scratch
    float*    fm8   = (float*)   alloc((size_t)512*1024*4);
    float*    gfeat = (float*)   alloc((size_t)2*1024*4);

    auto layer_gemm = [&](const float* fm, int ld, int li, float* outp,
                          int M, int N, int K) {
        wcvt_k<<<dim3((K*N + 255) / 256), 256, 0, stream>>>(Wl[li], whT, K, N);
        wmma_gemm_bias_k<<<dim3(N/32, M/16), 64, 0, stream>>>(fm, ld, whT, bl[li],
                                                              outp, N, K);
    };

    // ---- Level 1 (V=4096) ------------------------------------------------
    ball_query_k<<<dim3(16, 2), 256, 0, stream>>>(vertices, 4096, 0.04f, 32, idx1);
    conv_surface_k<<<dim3(1024), 256, 0, stream>>>(vertices, idx1, d0, fmA, 96, 4096);
    layer_gemm(fmA, 96, 1, gout, 8192, 160, 32);
    conv_agg_k<<<dim3(1024), 256, 0, stream>>>(vertices, idx1, gout, Dl[1],
                                               fmA, 96, 32, 4096, 1);
    layer_gemm(fmA, 96, 2, gout, 8192, 160, 64);
    conv_agg_k<<<dim3(1024), 256, 0, stream>>>(vertices, idx1, gout, Dl[2],
                                               fmA, 96, 64, 4096, 1);
    // pool 1: 4096 -> 1024, C=96
    ball_query_k<<<dim3(16, 2), 256, 0, stream>>>(vertices, 4096, 0.04f, 4, idxp);
    pool_select_k<<<dim3(16), 128, 0, stream>>>(vertices, fmA, 96, 96, idxp, 4096,
                                                vtx2, fmB, 192, 1024, 1877, 911, 4095);

    // ---- Level 2 (V=1024) ------------------------------------------------
    ball_query_k<<<dim3(4, 2), 256, 0, stream>>>(vtx2, 1024, 0.16f, 32, idx2);
    layer_gemm(fmB, 192, 3, gout, 2048, 160, 96);
    conv_agg_k<<<dim3(256), 256, 0, stream>>>(vtx2, idx2, gout, Dl[3],
                                              fmB, 192, 96, 1024, 1);
    layer_gemm(fmB, 192, 4, gout, 2048, 160, 128);
    conv_agg_k<<<dim3(256), 256, 0, stream>>>(vtx2, idx2, gout, Dl[4],
                                              fmB, 192, 128, 1024, 1);
    layer_gemm(fmB, 192, 5, gout, 2048, 160, 160);
    conv_agg_k<<<dim3(256), 256, 0, stream>>>(vtx2, idx2, gout, Dl[5],
                                              fmB, 192, 160, 1024, 1);
    // pool 2: 1024 -> 256, C=192
    ball_query_k<<<dim3(4, 2), 256, 0, stream>>>(vtx2, 1024, 0.16f, 4, idxp);
    pool_select_k<<<dim3(4), 128, 0, stream>>>(vtx2, fmB, 192, 192, idxp, 1024,
                                               vtx3, fmC, 256, 256, 733, 201, 1023);

    // ---- Level 3 (V=256) -------------------------------------------------
    ball_query_k<<<dim3(1, 2), 256, 0, stream>>>(vtx3, 256, 0.36f, 32, idx3);
    layer_gemm(fmC, 256, 6, gout, 512, 160, 192);
    conv_agg_k<<<dim3(64), 256, 0, stream>>>(vtx3, idx3, gout, Dl[6],
                                             fmC, 256, 192, 256, 1);
    layer_gemm(fmC, 256, 7, gout, 512, 160, 224);
    conv_agg_k<<<dim3(64), 256, 0, stream>>>(vtx3, idx3, gout, Dl[7],
                                             fmC, 256, 224, 256, 1);
    // layer 8: 512x5120x256 — the big WMMA GEMM
    layer_gemm(fmC, 256, 8, gout, 512, 5120, 256);
    conv_agg8_k<<<dim3(256, 2), 256, 0, stream>>>(vtx3, idx3, gout, Dl[8], fm8, 256);

    // ---- Head ------------------------------------------------------------
    gmax_k<<<dim3(8), 256, 0, stream>>>(fm8, gfeat);
    classifier_k<<<dim3(1), 256, 0, stream>>>(gfeat, cw1, cb1, bg, bb, cw2, cb2,
                                              (float*)d_out);
}
